// AttnBiasMHA_32418413150687
// MI455X (gfx1250) — matrix operand order repaired and checked by reading, not compiled, not run on hardware
//
#include <hip/hip_runtime.h>
#include <hip/hip_bf16.h>

typedef __bf16 bf16;
typedef __attribute__((ext_vector_type(16))) __bf16 v16bf;
typedef __attribute__((ext_vector_type(8)))  float  v8f;

#define T_DIM 1024
#define NF_DIM 512
#define H_DIM 8
#define DK_DIM 64
#define B_DIM 8

union FragU { v16bf v; unsigned u[8]; };

// A-fragment (16x32 bf16): lane r=lane&15 is row m, half=lane>>4 selects K group.
// VGPR v holds K pair: k = (v/4)*16 + half*8 + (v%4)*2   (ISA 7.12.2 table)
__device__ __forceinline__ v16bf load_a_frag(const bf16* rowptr, int half) {
  FragU f;
#pragma unroll
  for (int v = 0; v < 8; ++v) {
    int k = ((v >> 2) << 4) + (half << 3) + ((v & 3) << 1);
    f.u[v] = *reinterpret_cast<const unsigned*>(rowptr + k);
  }
  return f.v;
}

// B-fragment (32x16 bf16): lane r=lane&15 is col n; k = (v/4)*16 + half*8 + (v%4)*2 (same as A)
// colptr points at element (k=0, n=r) of a [n][k] (k-contiguous) layout.
__device__ __forceinline__ v16bf load_b_frag(const bf16* colptr, int half) {
  FragU f;
#pragma unroll
  for (int v = 0; v < 8; ++v) {
    int k = ((v >> 2) << 4) + (half << 3) + ((v & 3) << 1);
    f.u[v] = *reinterpret_cast<const unsigned*>(colptr + k);
  }
  return f.v;
}

__device__ __forceinline__ v8f wmma_bf16(v16bf a, v16bf b, v8f c) {
  return __builtin_amdgcn_wmma_f32_16x16x32_bf16(false, a, false, b, (short)0, c,
                                                 false, false);
}

// ---------------- LayerNorm -> bf16 -------------------------------------------
__global__ void __launch_bounds__(256) ln_to_bf16_k(const float* __restrict__ x,
                                                    const float* __restrict__ g,
                                                    const float* __restrict__ be,
                                                    bf16* __restrict__ out) {
  long row = blockIdx.x;
  const float* xr = x + row * NF_DIM;
  int tid = threadIdx.x;
  float v0 = xr[tid], v1 = xr[tid + 256];
  float s = v0 + v1, q = v0 * v0 + v1 * v1;
#pragma unroll
  for (int d = 1; d < 32; d <<= 1) {
    s += __shfl_xor(s, d, 32);
    q += __shfl_xor(q, d, 32);
  }
  __shared__ float ss[8], sq[8];
  int wv = tid >> 5, lane = tid & 31;
  if (lane == 0) { ss[wv] = s; sq[wv] = q; }
  __syncthreads();
  float ts = 0.f, tq = 0.f;
#pragma unroll
  for (int i = 0; i < 8; ++i) { ts += ss[i]; tq += sq[i]; }
  float mu = ts * (1.0f / NF_DIM);
  float var = tq * (1.0f / NF_DIM) - mu * mu;
  float rstd = rsqrtf(var + 1e-5f);
  out[row * NF_DIM + tid]       = (bf16)((v0 - mu) * rstd * g[tid] + be[tid]);
  out[row * NF_DIM + tid + 256] = (bf16)((v1 - mu) * rstd * g[tid + 256] + be[tid + 256]);
}

// ---------------- f32 -> bf16 convert -----------------------------------------
__global__ void cvt_f32_bf16_k(const float* __restrict__ in, bf16* __restrict__ out,
                               long n) {
  long i = (long)blockIdx.x * blockDim.x + threadIdx.x;
  long stride = (long)gridDim.x * blockDim.x;
  for (; i < n; i += stride) out[i] = (bf16)in[i];
}

// ---------------- W[K][N] f32 -> Wt[N][K] bf16 --------------------------------
__global__ void wtrans_k(const float* __restrict__ W, bf16* __restrict__ Wt,
                         int K, int N) {
  int i = blockIdx.x * 256 + threadIdx.x;  // i < K*N by grid sizing
  int k = i / N, n = i % N;
  Wt[(long)n * K + k] = (bf16)W[i];
}

// ---------------- V (B,T,H,DK) bf16 -> vT (B,H,DK,T) bf16 ---------------------
__global__ void vtrans_k(const bf16* __restrict__ v, bf16* __restrict__ vt) {
  int i = blockIdx.x * 256 + threadIdx.x;  // i < B*T*H*DK
  int d = i & 63, h = (i >> 6) & 7, t = (i >> 9) & 1023, b = i >> 19;
  vt[((((long)b * 8 + h) * 64 + d) << 10) + t] = v[i];
}

// ---------------- generic bf16 GEMM: C = A[M,K] @ Wt[N,K]^T + bias ------------
// one wave -> 16x64 output tile; block = 256 threads (8 waves)
__global__ void __launch_bounds__(256)
gemm_bf16_k(const bf16* __restrict__ A, const bf16* __restrict__ Wt,
            const float* __restrict__ bias, float* __restrict__ Cf,
            bf16* __restrict__ Cb, int M, int N, int K) {
  int lane = threadIdx.x & 31;
  int wave = blockIdx.x * 8 + (threadIdx.x >> 5);
  int nt4 = N >> 6;
  int tm = wave / nt4;
  int tn = wave % nt4;
  int half = lane >> 4, r = lane & 15;

  const bf16* arow = A + (long)(tm * 16 + r) * K;
  const bf16* brow[4];
#pragma unroll
  for (int j = 0; j < 4; ++j) brow[j] = Wt + (long)(tn * 64 + j * 16 + r) * K;

  v8f acc[4] = {};
  for (int kk = 0; kk < K; kk += 32) {
    __builtin_prefetch(arow + kk + 128, 0, 1);
    v16bf a = load_a_frag(arow + kk, half);
#pragma unroll
    for (int j = 0; j < 4; ++j)
      acc[j] = wmma_bf16(a, load_b_frag(brow[j] + kk, half), acc[j]);
  }
#pragma unroll
  for (int j = 0; j < 4; ++j) {
    int n = tn * 64 + j * 16 + r;
    float bv = bias[n];
#pragma unroll
    for (int v = 0; v < 8; ++v) {
      int m = tm * 16 + v + half * 8;
      float val = acc[j][v] + bv;
      if (Cf) Cf[(long)m * N + n] = val;
      else    Cb[(long)m * N + n] = (bf16)val;
    }
  }
}

// ---------------- positional bias: per t, (B*H=64 x DK) @ pos_k[t]^T ----------
// grid = T blocks, 8 waves; wave handles one bh-row-tile (of 4) x 32 s-tiles
__global__ void __launch_bounds__(256)
bias_wmma_k(const bf16* __restrict__ qb, const bf16* __restrict__ pk,
            float* __restrict__ bout) {
  int t = blockIdx.x;
  int lane = threadIdx.x & 31;
  int wv = threadIdx.x >> 5;
  int half = lane >> 4, r = lane & 15;
  int rt = wv & 3;       // bh tile 0..3
  int sh = wv >> 2;      // s half 0..1

  int bh = rt * 16 + r;  // row for A fragment
  const bf16* qrow =
      qb + (((long)(bh >> 3) * T_DIM + t) * H_DIM + (bh & 7)) * DK_DIM;
  v16bf a0 = load_a_frag(qrow, half);
  v16bf a1 = load_a_frag(qrow + 32, half);

  for (int st = sh * 32; st < sh * 32 + 32; ++st) {
    int s0 = st << 4;
    const bf16* prow = pk + ((long)t * T_DIM + s0 + r) * DK_DIM;
    v8f acc = {};
    acc = wmma_bf16(a0, load_b_frag(prow, half), acc);
    acc = wmma_bf16(a1, load_b_frag(prow + 32, half), acc);
#pragma unroll
    for (int v = 0; v < 8; ++v) {
      int bhr = rt * 16 + v + half * 8;
      bout[((long)bhr << 20) + ((long)t << 10) + s0 + r] = acc[v];
    }
  }
}

// ---------------- flash attention with precomputed bias -----------------------
// one wave per (b, h, 16-row t tile); block = 128 (4 waves)
__global__ void __launch_bounds__(128)
attn_flash_k(const bf16* __restrict__ qb, const bf16* __restrict__ kb,
             const bf16* __restrict__ vt, const float* __restrict__ biasB,
             const int* __restrict__ mask, bf16* __restrict__ aout) {
  __shared__ __align__(16) bf16 ldsP[4][16 * 32];
  int lane = threadIdx.x & 31;
  int wv = threadIdx.x >> 5;
  int gw = blockIdx.x * 4 + wv;
  int t0 = (gw & 63) << 4;
  int h = (gw >> 6) & 7;
  int b = gw >> 9;
  int half = lane >> 4, r = lane & 15;
  long bh = (long)(b * 8 + h);

  const bf16* qrow = qb + (((long)b * T_DIM + t0 + r) * H_DIM + h) * DK_DIM;
  v16bf aq0 = load_a_frag(qrow, half);
  v16bf aq1 = load_a_frag(qrow + 32, half);

  float mrow[8], lrow[8];
  v8f o[4] = {};
#pragma unroll
  for (int v = 0; v < 8; ++v) { mrow[v] = -3.4e38f; lrow[v] = 0.f; }

  bf16* L = &ldsP[wv][0];

  for (int sb = 0; sb < 32; ++sb) {   // 32 blocks of 32 columns
    int s0 = sb << 5;
    v8f sc[2];
#pragma unroll
    for (int u = 0; u < 2; ++u) {
      const bf16* krow =
          kb + (((long)b * T_DIM + s0 + u * 16 + r) * H_DIM + h) * DK_DIM;
      v8f a = {};
      a = wmma_bf16(aq0, load_b_frag(krow, half), a);
      a = wmma_bf16(aq1, load_b_frag(krow + 32, half), a);
      sc[u] = a;
    }

    float p[2][8];
    v8f av = {};
#pragma unroll
    for (int v = 0; v < 8; ++v) {
      int t = t0 + v + half * 8;
      long boff = (bh << 20) + ((long)t << 10) + s0 + r;
      long moff = ((long)b << 20) + ((long)t << 10) + s0 + r;
      float sv[2];
      float mx = -3.4e38f;
#pragma unroll
      for (int u = 0; u < 2; ++u) {
        float bias = biasB[boff + u * 16];
        int mk = mask[moff + u * 16];
        float s = (sc[u][v] + bias) * 0.125f;     // /sqrt(DK)
        sv[u] = (mk == 0) ? -3.4e38f : s;
        mx = fmaxf(mx, sv[u]);
      }
#pragma unroll
      for (int d = 1; d < 16; d <<= 1) mx = fmaxf(mx, __shfl_xor(mx, d, 32));
      float mnew = fmaxf(mrow[v], mx);
      float alpha = __expf(mrow[v] - mnew);
      float rs = 0.f;
#pragma unroll
      for (int u = 0; u < 2; ++u) {
        float pv = (sv[u] <= -3.0e38f) ? 0.f : __expf(sv[u] - mnew);
        p[u][v] = pv;
        rs += pv;
      }
#pragma unroll
      for (int d = 1; d < 16; d <<= 1) rs += __shfl_xor(rs, d, 32);
      lrow[v] = lrow[v] * alpha + rs;
      mrow[v] = mnew;
      av[v] = alpha;
    }
#pragma unroll
    for (int j = 0; j < 4; ++j) o[j] = o[j] * av;

    // stage P (C layout) -> LDS -> reload as A fragment (16x32 bf16)
#pragma unroll
    for (int v = 0; v < 8; ++v) {
      int m = v + half * 8;
      L[m * 32 + r]      = (bf16)p[0][v];
      L[m * 32 + 16 + r] = (bf16)p[1][v];
    }
    asm volatile("s_wait_dscnt 0x0" ::: "memory");  // cross-lane LDS visibility
    v16bf pa = load_a_frag(L + r * 32, half);

    const bf16* vbase = vt + (bh * DK_DIM + r) * T_DIM + s0;
#pragma unroll
    for (int j = 0; j < 4; ++j)
      o[j] = wmma_bf16(pa, load_b_frag(vbase + (long)j * 16 * T_DIM, half), o[j]);
  }

#pragma unroll
  for (int j = 0; j < 4; ++j) {
#pragma unroll
    for (int v = 0; v < 8; ++v) {
      int t = t0 + v + half * 8;
      float l = lrow[v];
      float val = (l > 0.f) ? o[j][v] / l : 0.f;
      aout[((long)b * T_DIM + t) * NF_DIM + h * DK_DIM + j * 16 + r] = (bf16)val;
    }
  }
}

// ------------------------------------------------------------------------------
extern "C" void kernel_launch(void* const* d_in, const int* in_sizes, int n_in,
                              void* d_out, int out_size, void* d_ws, size_t ws_size,
                              hipStream_t stream) {
  (void)in_sizes; (void)n_in; (void)out_size; (void)ws_size;
  const float* x     = (const float*)d_in[0];
  const float* cond  = (const float*)d_in[1];
  const float* pos_k = (const float*)d_in[2];
  const int*   mask  = (const int*)d_in[3];
  const float* lng   = (const float*)d_in[4];
  const float* lnb   = (const float*)d_in[5];
  const float* Wq = (const float*)d_in[6];  const float* bq = (const float*)d_in[7];
  const float* Wk = (const float*)d_in[8];  const float* bk = (const float*)d_in[9];
  const float* Wv = (const float*)d_in[10]; const float* bv = (const float*)d_in[11];
  const float* Wo = (const float*)d_in[12]; const float* bo = (const float*)d_in[13];
  const float* Wc = (const float*)d_in[14]; const float* bc = (const float*)d_in[15];
  float* out = (float*)d_out;

  const size_t SZ_PK   = (size_t)T_DIM * T_DIM * DK_DIM * 2;   // 128 MB
  const size_t SZ_ACT  = (size_t)B_DIM * T_DIM * NF_DIM * 2;   // 8 MB
  const size_t SZ_COND = (size_t)B_DIM * T_DIM * 256 * 2;      // 4 MB
  const size_t SZ_WCT  = (size_t)512 * 256 * 2;
  const size_t SZ_WT   = (size_t)512 * 512 * 2;
  const size_t SZ_BIAS = (size_t)64 * T_DIM * T_DIM * 4;       // 256 MB

  char* w = (char*)d_ws;
  bf16*  pk_bf   = (bf16*)w;  w += SZ_PK;
  bf16*  xn_bf   = (bf16*)w;  w += SZ_ACT;
  bf16*  cond_bf = (bf16*)w;  w += SZ_COND;
  bf16*  tq_bf   = (bf16*)w;  w += SZ_ACT;
  bf16*  q_bf    = (bf16*)w;  w += SZ_ACT;
  bf16*  k_bf    = (bf16*)w;  w += SZ_ACT;
  bf16*  v_bf    = (bf16*)w;  w += SZ_ACT;
  bf16*  vT_bf   = (bf16*)w;  w += SZ_ACT;
  bf16*  ao_bf   = (bf16*)w;  w += SZ_ACT;
  bf16*  Wct     = (bf16*)w;  w += SZ_WCT;
  bf16*  Wqt     = (bf16*)w;  w += SZ_WT;
  bf16*  Wkt     = (bf16*)w;  w += SZ_WT;
  bf16*  Wvt     = (bf16*)w;  w += SZ_WT;
  bf16*  Wot     = (bf16*)w;  w += SZ_WT;
  float* bias_f  = (float*)w; w += SZ_BIAS;

  // --- precision/layout conversions ---
  cvt_f32_bf16_k<<<4096, 256, 0, stream>>>(pos_k, pk_bf, (long)T_DIM * T_DIM * DK_DIM);
  cvt_f32_bf16_k<<<2048, 256, 0, stream>>>(cond, cond_bf, (long)B_DIM * T_DIM * 256);
  ln_to_bf16_k<<<B_DIM * T_DIM, 256, 0, stream>>>(x, lng, lnb, xn_bf);
  wtrans_k<<<512, 256, 0, stream>>>(Wc, Wct, 256, 512);
  wtrans_k<<<1024, 256, 0, stream>>>(Wq, Wqt, 512, 512);
  wtrans_k<<<1024, 256, 0, stream>>>(Wk, Wkt, 512, 512);
  wtrans_k<<<1024, 256, 0, stream>>>(Wv, Wvt, 512, 512);
  wtrans_k<<<1024, 256, 0, stream>>>(Wo, Wot, 512, 512);

  // --- projections (M=8192; waves = (M/16)*(N/64) = 4096 -> 512 blocks) ---
  gemm_bf16_k<<<512, 256, 0, stream>>>(cond_bf, Wct, bc, nullptr, tq_bf, 8192, 512, 256);
  gemm_bf16_k<<<512, 256, 0, stream>>>(tq_bf,   Wqt, bq, nullptr, q_bf,  8192, 512, 512);
  gemm_bf16_k<<<512, 256, 0, stream>>>(xn_bf,   Wkt, bk, nullptr, k_bf,  8192, 512, 512);
  gemm_bf16_k<<<512, 256, 0, stream>>>(xn_bf,   Wvt, bv, nullptr, v_bf,  8192, 512, 512);

  vtrans_k<<<16384, 256, 0, stream>>>(v_bf, vT_bf);

  // --- positional bias + attention ---
  bias_wmma_k<<<T_DIM, 256, 0, stream>>>(q_bf, pk_bf, bias_f);
  attn_flash_k<<<1024, 128, 0, stream>>>(q_bf, k_bf, vT_bf, bias_f, mask, ao_bf);

  // --- output projection (f32 result) ---
  gemm_bf16_k<<<512, 256, 0, stream>>>(ao_bf, Wot, bo, out, nullptr, 8192, 512, 512);
}
